// TransformerDecoderBlock_11012296147295
// MI455X (gfx1250) — matrix operand order repaired and checked by reading, not compiled, not run on hardware
//
#include <hip/hip_runtime.h>
#include <hip/hip_bf16.h>
#include <math.h>

// ---------------------------------------------------------------------------
// Problem constants (from the reference)
// ---------------------------------------------------------------------------
#define Bb 4
#define Ss 4096
#define Dd 512
#define Ff 2048

typedef __bf16 bf16;
typedef __attribute__((ext_vector_type(16))) __bf16 v16bf;
typedef __attribute__((ext_vector_type(8)))  float   v8f;
typedef __attribute__((ext_vector_type(4)))  unsigned int v4u;
typedef __attribute__((ext_vector_type(8)))  int     v8i;
typedef __attribute__((ext_vector_type(4)))  int     v4i;

// ---------------------------------------------------------------------------
// Tensor Data Mover staging (CDNA5): one wave posts a 2-D tile descriptor,
// the TDM streams global->LDS with no VGPR traffic; sync via TENSORcnt.
// ---------------------------------------------------------------------------
#if defined(__HIP_DEVICE_COMPILE__) && __has_builtin(__builtin_amdgcn_tensor_load_to_lds)
#define USE_TDM 1
#else
#define USE_TDM 0
#endif

#if USE_TDM
// Load a dense tile of `rows` x `dim0` bf16 elements (row stride `stride`
// elements in global memory) into LDS at byte offset ldsAddr.
__device__ inline void tdm_load_2d(unsigned ldsAddr, const void* gptr,
                                   unsigned dim0, unsigned rows, unsigned stride)
{
    const unsigned long long ga = (unsigned long long)gptr;
    // ---- D# group 0: control / lds_addr / global_addr / type ----
    v4u g0;
    g0[0] = 1u;                                    // count=1, no gather, user desc
    g0[1] = ldsAddr;                               // lds_addr (bytes)
    g0[2] = (unsigned)(ga & 0xffffffffu);          // global_addr[31:0]
    g0[3] = (unsigned)((ga >> 32) & 0x01ffffffu)   // global_addr[56:32]
          | (2u << 30);                            // type = 2 ("image")
    // ---- D# group 1: data_size / dims / tile dims / dim0 stride ----
    v8i g1;
    g1[0] = (int)(1u << 16);                       // workgroup_mask=0, data_size=1 (2 bytes)
    g1[1] = (int)((dim0 & 0xffffu) << 16);         // tensor_dim0[15:0] @ bits[63:48]
    g1[2] = (int)(((dim0 >> 16) & 0xffffu)         // tensor_dim0[31:16]
          | ((rows & 0xffffu) << 16));             // tensor_dim1[15:0]
    g1[3] = (int)(((rows >> 16) & 0xffffu)         // tensor_dim1[31:16]
          | ((dim0 & 0xffffu) << 16));             // tile_dim0
    g1[4] = (int)(rows & 0xffffu);                 // tile_dim1 (tile_dim2=0)
    g1[5] = (int)stride;                           // tensor_dim0_stride[31:0]
    g1[6] = 0;                                     // stride hi / dim1_stride lo
    g1[7] = 0;
    v4i gz;
    gz[0] = 0; gz[1] = 0; gz[2] = 0; gz[3] = 0;    // groups 2/3 unused (2-D tile)
#if defined(__clang_major__) && (__clang_major__ >= 23)
    v8i g5;
    for (int i = 0; i < 8; ++i) g5[i] = 0;
    __builtin_amdgcn_tensor_load_to_lds(g0, g1, gz, gz, g5, 0);
#else
    __builtin_amdgcn_tensor_load_to_lds(g0, g1, gz, gz, 0);
#endif
}
#endif

// Stage a rows x dim0 bf16 tile into (dense) LDS. Caller brackets with
// __syncthreads(). With TDM: wave 0 posts the descriptor and drains TENSORcnt.
__device__ inline void stage_tile(bf16* lds, const bf16* src,
                                  unsigned dim0, unsigned rows, unsigned stride)
{
#if USE_TDM
    if ((threadIdx.x >> 5) == 0) {
        tdm_load_2d((unsigned)(size_t)lds, (const void*)src, dim0, rows, stride);
        __builtin_amdgcn_s_wait_tensorcnt((short)0);
    }
#else
    const int row_u4 = (int)dim0 / 8;
    const int n_u4   = (int)rows * row_u4;
    for (int i = threadIdx.x; i < n_u4; i += 256) {
        const int r = i / row_u4, c = i % row_u4;
        reinterpret_cast<uint4*>(lds + (size_t)r * dim0)[c] =
            reinterpret_cast<const uint4*>(src + (size_t)r * stride)[c];
    }
#endif
}

// ---------------------------------------------------------------------------
// WMMA helpers (CDNA5: D = A(16x32 bf16) * B(32x16 bf16) + C(16x16 f32))
// A and N-major B fragment loads are contiguous-per-lane (32B runs -> ds_load_b128 pairs),
// per ISA §10.9: 16-bit WMMA operands are fed with 128-bit loads.
// ---------------------------------------------------------------------------
__device__ inline v8f wmma_bf16(v16bf a, v16bf b, v8f c) {
    return __builtin_amdgcn_wmma_f32_16x16x32_bf16(
        /*neg_a=*/false, a, /*neg_b=*/false, b,
        /*c_mod=*/(short)0, c, /*reuse_a=*/false, /*reuse_b=*/false);
}

__device__ inline v8f zero_v8f() {
    v8f z;
#pragma unroll
    for (int i = 0; i < 8; ++i) z[i] = 0.0f;
    return z;
}

// A-fragment: logical A[m][k], m in [0,16), k in [0,32), src row-major with ld.
// One contiguous 16-element (32B) run per lane: k = 16*(lane/16) + e.
__device__ inline v16bf load_a_frag(const bf16* src, int ld) {
    int lane  = threadIdx.x & 31;
    int m     = lane & 15;
    int kbase = (lane >> 4) << 4;   // 0 or 16
    v16bf a;
#pragma unroll
    for (int e = 0; e < 16; ++e) {
        int k = kbase + e;
        a[e] = src[m * ld + k];
    }
    return a;
}

// B-fragment from an N-major source: B[k][n] = src[n*ld + k].
// One contiguous 16-element (32B) run per lane.
__device__ inline v16bf load_bt_frag(const bf16* src, int ld) {
    int lane  = threadIdx.x & 31;
    int n     = lane & 15;
    int kbase = (lane >> 4) << 4;   // 0 or 16
    v16bf b;
#pragma unroll
    for (int e = 0; e < 16; ++e) {
        b[e] = src[n * ld + kbase + e];
    }
    return b;
}

// B-fragment from a row-major [32 x >=16] tile: n = lane%16,
// k = 16*(lane/16) + e (strided along k-rows).
__device__ inline v16bf load_b_rm_frag(const bf16* src, int ld) {
    int lane = threadIdx.x & 31;
    int n     = lane & 15;
    int kbase = (lane >> 4) << 4;
    v16bf b;
#pragma unroll
    for (int e = 0; e < 16; ++e) {
        b[e] = src[(kbase + e) * ld + n];
    }
    return b;
}

// ---------------------------------------------------------------------------
// Flash attention: out[b, q, :] = softmax(Q Kᵀ * scale [causal]) V
// Block = 256 threads (8 waves), 16 query rows, full D=512.
// Waves split the feature dim (64 each); score partials reduced via LDS.
// K/V tiles streamed by the Tensor Data Mover.
// ---------------------------------------------------------------------------
#define QT 16
#define KT 32

__global__ __launch_bounds__(256)
void flash_attn_kernel(const bf16* __restrict__ Q, const bf16* __restrict__ K,
                       const bf16* __restrict__ V, float* __restrict__ O,
                       int causal, float scale)
{
    __shared__ __align__(16) bf16  ldsKV[KT * Dd];       // 32KB (K, then V; also Q staging)
    __shared__ float ldsPart[8][512];                    // 16KB score partials
    __shared__ float ldsS[QT * KT];                      // 2KB reduced scores
    __shared__ __align__(16) bf16  ldsP[QT * KT];        // 1KB probabilities
    __shared__ float ldsAlpha[QT];
    __shared__ float ldsM[QT];
    __shared__ float ldsL[QT];

    const int b    = blockIdx.y;
    const int q0   = blockIdx.x * QT;
    const int tid  = threadIdx.x;
    const int wave = tid >> 5;
    const int lane = tid & 31;

    const size_t base = (size_t)b * Ss * Dd;

    // Stage Q tile (16 contiguous rows) and pull per-wave A-fragments to VGPRs.
    stage_tile(ldsKV, Q + base + (size_t)q0 * Dd, Dd, QT, Dd);
    __syncthreads();
    v16bf qfrag[2];
#pragma unroll
    for (int c = 0; c < 2; ++c)
        qfrag[c] = load_a_frag(ldsKV + wave * 64 + c * 32, Dd);
    __syncthreads();

    if (tid < QT) { ldsM[tid] = -3.0e38f; ldsL[tid] = 0.0f; }

    v8f acc[4];
#pragma unroll
    for (int t = 0; t < 4; ++t) acc[t] = zero_v8f();

    const int nkb = causal ? ((q0 + QT + KT - 1) / KT) : (Ss / KT);

    for (int kb = 0; kb < nkb; ++kb) {
        const int k0 = kb * KT;
        __syncthreads();
        stage_tile(ldsKV, K + base + (size_t)k0 * Dd, Dd, KT, Dd);
        __syncthreads();

        // --- partial scores over this wave's 64 features -------------------
        v8f sp[2];
        sp[0] = zero_v8f(); sp[1] = zero_v8f();
#pragma unroll
        for (int st = 0; st < 2; ++st) {
#pragma unroll
            for (int c = 0; c < 2; ++c) {
                v16bf kf = load_bt_frag(ldsKV + (st * 16) * Dd + wave * 64 + c * 32, Dd);
                sp[st] = wmma_bf16(qfrag[c], kf, sp[st]);
            }
        }
#pragma unroll
        for (int st = 0; st < 2; ++st)
#pragma unroll
            for (int j = 0; j < 8; ++j)
                ldsPart[wave][st * 256 + j * 32 + lane] = sp[st][j];
        __syncthreads();

        // --- cross-wave reduce + scale + causal mask -----------------------
        for (int e = tid; e < 512; e += 256) {
            float s = 0.0f;
#pragma unroll
            for (int w = 0; w < 8; ++w) s += ldsPart[w][e];
            const int t  = e >> 8;
            const int j  = (e >> 5) & 7;
            const int ln = e & 31;
            const int M  = j + ((ln >= 16) ? 8 : 0);
            const int N  = (ln & 15) + t * 16;
            s *= scale;
            if (causal && (k0 + N > q0 + M)) s = -1.0e30f;
            ldsS[M * KT + N] = s;
        }
        __syncthreads();

        // --- online softmax (one thread per query row) ---------------------
        if (tid < QT) {
            const int M = tid;
            float mold = ldsM[M];
            float mnew = mold;
            for (int N = 0; N < KT; ++N) mnew = fmaxf(mnew, ldsS[M * KT + N]);
            const float alpha = __expf(mold - mnew);
            float lsum = 0.0f;
            for (int N = 0; N < KT; ++N) {
                float p = __expf(ldsS[M * KT + N] - mnew);
                ldsP[M * KT + N] = (bf16)p;
                lsum += p;
            }
            ldsM[M]     = mnew;
            ldsL[M]     = ldsL[M] * alpha + lsum;
            ldsAlpha[M] = alpha;
        }
        __syncthreads();

        // --- rescale accumulators ------------------------------------------
        {
            const int moff = (lane >= 16) ? 8 : 0;
#pragma unroll
            for (int j = 0; j < 8; ++j) {
                const float a = ldsAlpha[j + moff];
#pragma unroll
                for (int t = 0; t < 4; ++t) acc[t][j] *= a;
            }
        }
        __syncthreads();

        // --- stream V block over the same buffer; accumulate P*V -----------
        stage_tile(ldsKV, V + base + (size_t)k0 * Dd, Dd, KT, Dd);
        __syncthreads();
        v16bf pf = load_a_frag(ldsP, KT);
#pragma unroll
        for (int t = 0; t < 4; ++t) {
            // V tile is row-major [32 keys x 512 feats]; lane%16 = feature col,
            // 16 key rows k = 16*(lane/16)+e, strided by Dd per lane.
            v16bf vf = load_b_rm_frag(ldsKV + wave * 64 + t * 16, Dd);
            acc[t] = wmma_bf16(pf, vf, acc[t]);
        }
    }

    // --- normalize and write 16x64 slice -----------------------------------
    const int moff = (lane >= 16) ? 8 : 0;
#pragma unroll
    for (int t = 0; t < 4; ++t) {
#pragma unroll
        for (int j = 0; j < 8; ++j) {
            const int M   = j + moff;
            const int col = wave * 64 + t * 16 + (lane & 15);
            O[base + (size_t)(q0 + M) * Dd + col] = acc[t][j] / ldsL[M];
        }
    }
}

// ---------------------------------------------------------------------------
// GEMM: C[m,n] = sum_k A[m,k] * W[n,k] (+ bias[n]) (+ res[m,n]) (ReLU?)
// A bf16 [M x K], W bf16 [N x K] row-major (the reference's natural layout,
// i.e. X @ W.T with no pre-transpose). Block tile 32x128; A/W tiles streamed
// by the Tensor Data Mover; W tile is N-major so B-fragments are contiguous.
// ---------------------------------------------------------------------------
__global__ __launch_bounds__(256)
void gemm_bf16_kernel(const bf16* __restrict__ A, const bf16* __restrict__ W,
                      const float* __restrict__ bias, const float* __restrict__ res,
                      float* __restrict__ Cf, bf16* __restrict__ Cb,
                      int M, int N, int Kd, int relu)
{
    __shared__ __align__(16) bf16 ldsA[32 * 32];    // 2KB   [32 m][32 k]
    __shared__ __align__(16) bf16 ldsB[128 * 32];   // 8KB   [128 n][32 k]

    const int m0   = blockIdx.x * 32;
    const int n0   = blockIdx.y * 128;
    const int tid  = threadIdx.x;
    const int wave = tid >> 5;
    const int lane = tid & 31;
    const int mw   = (wave >> 2) * 16;   // 0 / 16
    const int nw   = (wave & 3) * 32;    // 0 / 32 / 64 / 96

    v8f acc[2];
    acc[0] = zero_v8f(); acc[1] = zero_v8f();

    for (int k0 = 0; k0 < Kd; k0 += 32) {
        __syncthreads();
        stage_tile(ldsA, A + (size_t)m0 * Kd + k0, 32, 32, Kd);
        stage_tile(ldsB, W + (size_t)n0 * Kd + k0, 32, 128, Kd);
        __syncthreads();

        v16bf af = load_a_frag(ldsA + mw * 32, 32);
#pragma unroll
        for (int t = 0; t < 2; ++t) {
            v16bf bf_ = load_bt_frag(ldsB + (nw + t * 16) * 32, 32);
            acc[t] = wmma_bf16(af, bf_, acc[t]);
        }
    }

    const int moff = (lane >= 16) ? 8 : 0;
#pragma unroll
    for (int t = 0; t < 2; ++t) {
#pragma unroll
        for (int j = 0; j < 8; ++j) {
            const int row = m0 + mw + j + moff;
            const int col = n0 + nw + t * 16 + (lane & 15);
            float v = acc[t][j];
            if (bias) v += bias[col];
            if (res)  v += res[(size_t)row * N + col];
            if (relu) v = fmaxf(v, 0.0f);
            if (Cf) Cf[(size_t)row * N + col] = v;
            if (Cb) Cb[(size_t)row * N + col] = (bf16)v;
        }
    }
}

// ---------------------------------------------------------------------------
// LayerNorm over (S,D) jointly, per batch sample. Deterministic 2-stage reduce.
// ---------------------------------------------------------------------------
__global__ __launch_bounds__(256)
void ln_partial_kernel(const float* __restrict__ x, float* __restrict__ part,
                       int perB, int nblk)
{
    __shared__ float sh[256], sh2[256];
    const int b = blockIdx.y;
    const float* xb = x + (size_t)b * perB;
    float s = 0.0f, s2 = 0.0f;
    for (int i = blockIdx.x * 256 + threadIdx.x; i < perB; i += nblk * 256) {
        const float v = xb[i];
        s += v; s2 += v * v;
    }
    sh[threadIdx.x] = s; sh2[threadIdx.x] = s2;
    __syncthreads();
    for (int o = 128; o > 0; o >>= 1) {
        if (threadIdx.x < o) { sh[threadIdx.x] += sh[threadIdx.x + o]; sh2[threadIdx.x] += sh2[threadIdx.x + o]; }
        __syncthreads();
    }
    if (threadIdx.x == 0) {
        part[((size_t)b * nblk + blockIdx.x) * 2]     = sh[0];
        part[((size_t)b * nblk + blockIdx.x) * 2 + 1] = sh2[0];
    }
}

__global__ __launch_bounds__(256)
void ln_finalize_kernel(const float* __restrict__ part, float* __restrict__ stats,
                        int nblk, float inv_n)
{
    __shared__ float sh[256], sh2[256];
    const int b = blockIdx.x, tid = threadIdx.x;
    float s = 0.0f, s2 = 0.0f;
    for (int i = tid; i < nblk; i += 256) {
        s  += part[((size_t)b * nblk + i) * 2];
        s2 += part[((size_t)b * nblk + i) * 2 + 1];
    }
    sh[tid] = s; sh2[tid] = s2;
    __syncthreads();
    for (int o = 128; o > 0; o >>= 1) {
        if (tid < o) { sh[tid] += sh[tid + o]; sh2[tid] += sh2[tid + o]; }
        __syncthreads();
    }
    if (tid == 0) {
        const float mu  = sh[0] * inv_n;
        const float var = sh2[0] * inv_n - mu * mu;
        stats[b * 2]     = mu;
        stats[b * 2 + 1] = rsqrtf(var + 1e-5f);
    }
}

__global__ __launch_bounds__(256)
void ln_apply_kernel(const float* __restrict__ x, const float* __restrict__ g,
                     const float* __restrict__ bta, const float* __restrict__ stats,
                     float* __restrict__ y, bf16* __restrict__ yb, int perB)
{
    const size_t idx = (size_t)blockIdx.x * 256 + threadIdx.x;
    const int b = (int)(idx / perB);
    const int r = (int)(idx % perB);
    const float mu   = stats[b * 2];
    const float rstd = stats[b * 2 + 1];
    const float v = (x[idx] - mu) * rstd * g[r] + bta[r];
    if (y)  y[idx]  = v;
    if (yb) yb[idx] = (bf16)v;
}

// ---------------------------------------------------------------------------
// fp32 -> bf16 cast (contiguous, n % 1024 == 0 for all users)
// ---------------------------------------------------------------------------
__global__ __launch_bounds__(256)
void cast_f32_bf16_kernel(const float* __restrict__ x, bf16* __restrict__ y)
{
    const size_t i = ((size_t)blockIdx.x * 256 + threadIdx.x) * 4;
    const float4 v = *reinterpret_cast<const float4*>(x + i);
    y[i]     = (bf16)v.x;
    y[i + 1] = (bf16)v.y;
    y[i + 2] = (bf16)v.z;
    y[i + 3] = (bf16)v.w;
}

// ---------------------------------------------------------------------------
// Host-side orchestration
// ---------------------------------------------------------------------------
extern "C" void kernel_launch(void* const* d_in, const int* in_sizes, int n_in,
                              void* d_out, int out_size, void* d_ws, size_t ws_size,
                              hipStream_t stream)
{
    (void)in_sizes; (void)n_in; (void)out_size; (void)ws_size;

    const float* x_in  = (const float*)d_in[0];
    const float* enc   = (const float*)d_in[1];
    const float* Wo1   = (const float*)d_in[2];
    const float* Wo2   = (const float*)d_in[3];
    const float* ln1_g = (const float*)d_in[4];
    const float* ln1_b = (const float*)d_in[5];
    const float* ln2_g = (const float*)d_in[6];
    const float* ln2_b = (const float*)d_in[7];
    const float* W1    = (const float*)d_in[8];
    const float* b1    = (const float*)d_in[9];
    const float* W2    = (const float*)d_in[10];
    const float* b2    = (const float*)d_in[11];
    float* out = (float*)d_out;

    const size_t P  = (size_t)Bb * Ss * Dd;   // 8388608
    const size_t PF = (size_t)Bb * Ss * Ff;   // 33554432
    const int    perB = Ss * Dd;
    const int    Mrows = Bb * Ss;             // 16384
    const float  scale = 0.044194173824159216f; // 1/sqrt(512)

    char* wp = (char*)d_ws;
    auto carve = [&](size_t bytes) -> void* {
        void* p = (void*)wp;
        wp += (bytes + 255) & ~(size_t)255;
        return p;
    };
    float* T0    = (float*)carve(P * 4);
    float* T1    = (float*)carve(P * 4);
    bf16*  xb    = (bf16*)carve(P * 2);
    bf16*  encb  = (bf16*)carve(P * 2);
    bf16*  castb = (bf16*)carve(P * 2);
    bf16*  lnb   = (bf16*)carve(P * 2);
    bf16*  hidb  = (bf16*)carve(PF * 2);
    bf16*  wo1b  = (bf16*)carve((size_t)Dd * Dd * 2);
    bf16*  wo2b  = (bf16*)carve((size_t)Dd * Dd * 2);
    bf16*  w1b   = (bf16*)carve((size_t)Ff * Dd * 2);
    bf16*  w2b   = (bf16*)carve((size_t)Dd * Ff * 2);
    float* part  = (float*)carve((size_t)Bb * 256 * 2 * 4);
    float* stats = (float*)carve((size_t)Bb * 2 * 4);

    const dim3 blk(256);
    const dim3 gCast((unsigned)(P / 4 / 256));
    const dim3 gAttn(Ss / QT, Bb);
    const dim3 gProj(Mrows / 32, Dd / 128);
    const dim3 gFfn1(Mrows / 32, Ff / 128);
    const dim3 gLnP(256, Bb);
    const dim3 gLnA((unsigned)(P / 256));
    const unsigned gWdd = (unsigned)((size_t)Dd * Dd / 4 / 256);
    const unsigned gWfd = (unsigned)((size_t)Ff * Dd / 4 / 256);

    // ---- precompute bf16 operands (weights keep their natural [N,K] layout) ----
    cast_f32_bf16_kernel<<<gCast, blk, 0, stream>>>(x_in, xb);
    cast_f32_bf16_kernel<<<gCast, blk, 0, stream>>>(enc, encb);
    cast_f32_bf16_kernel<<<gWdd, blk, 0, stream>>>(Wo1, wo1b);
    cast_f32_bf16_kernel<<<gWdd, blk, 0, stream>>>(Wo2, wo2b);
    cast_f32_bf16_kernel<<<gWfd, blk, 0, stream>>>(W1, w1b);
    cast_f32_bf16_kernel<<<gWfd, blk, 0, stream>>>(W2, w2b);

    // ---- stage 1: causal self-attention + Wo1 + residual + LN1 ----
    flash_attn_kernel<<<gAttn, blk, 0, stream>>>(xb, xb, xb, T0, 1, scale);
    cast_f32_bf16_kernel<<<gCast, blk, 0, stream>>>(T0, castb);
    gemm_bf16_kernel<<<gProj, blk, 0, stream>>>(castb, wo1b, nullptr, x_in,
                                                T1, nullptr, Mrows, Dd, Dd, 0);
    ln_partial_kernel<<<gLnP, blk, 0, stream>>>(T1, part, perB, 256);
    ln_finalize_kernel<<<Bb, blk, 0, stream>>>(part, stats, 256, 1.0f / perB);
    ln_apply_kernel<<<gLnA, blk, 0, stream>>>(T1, ln1_g, ln1_b, stats, T0, lnb, perB);

    // ---- stage 2: cross attention (q=enc,k=enc,v=x) + Wo2 + residual + LN2 ----
    flash_attn_kernel<<<gAttn, blk, 0, stream>>>(encb, encb, lnb, T1, 0, scale);
    cast_f32_bf16_kernel<<<gCast, blk, 0, stream>>>(T1, castb);
    gemm_bf16_kernel<<<gProj, blk, 0, stream>>>(castb, wo2b, nullptr, T0,
                                                T1, nullptr, Mrows, Dd, Dd, 0);
    ln_partial_kernel<<<gLnP, blk, 0, stream>>>(T1, part, perB, 256);
    ln_finalize_kernel<<<Bb, blk, 0, stream>>>(part, stats, 256, 1.0f / perB);
    ln_apply_kernel<<<gLnA, blk, 0, stream>>>(T1, ln2_g, ln2_b, stats, T0, lnb, perB);

    // ---- stage 3: FFN + residual + LN2 (shared params) ----
    gemm_bf16_kernel<<<gFfn1, blk, 0, stream>>>(lnb, w1b, b1, nullptr,
                                                nullptr, hidb, Mrows, Ff, Dd, 1);
    gemm_bf16_kernel<<<gProj, blk, 0, stream>>>(hidb, w2b, b2, T0,
                                                T1, nullptr, Mrows, Dd, Ff, 0);
    ln_partial_kernel<<<gLnP, blk, 0, stream>>>(T1, part, perB, 256);
    ln_finalize_kernel<<<Bb, blk, 0, stream>>>(part, stats, 256, 1.0f / perB);
    ln_apply_kernel<<<gLnA, blk, 0, stream>>>(T1, ln2_g, ln2_b, stats, out, nullptr, perB);
}
